// IFFTv2_19490561589945
// MI455X (gfx1250) — matrix operand order repaired and checked by reading, not compiled, not run on hardware
//
#include <hip/hip_runtime.h>

typedef __attribute__((ext_vector_type(16))) __bf16        v16bf;
typedef __attribute__((ext_vector_type(8)))  __bf16        v8bf;
typedef __attribute__((ext_vector_type(8)))  float         v8f;
typedef __attribute__((ext_vector_type(8)))  unsigned int  v8u;

#define N_FFT   1024
#define N_SUB   128
#define N_SLOT  8
#define TM      16      // batch rows per workgroup
#define THREADS 256     // 8 waves (wave32)

__device__ __forceinline__ int bitrev3(int x) {
    return ((x & 1) << 2) | (x & 2) | ((x >> 2) & 1);
}

// ---------------------------------------------------------------------------
// Pre-pass: pack the four 128x128 fp32 IDFT matrices into bf16 B-matrix
// fragments in d_ws.  Fragment (kc, nt) holds B[k][n] = W[n][k] for
// n = nt*16 + (lane&15), k = kc*32 + (lane>>4)*8 + (j&7) + (j>>3)*16  (matching
// the CDNA5 16-bit B-matrix 32x16 VGPR layout).  Layout in ws:
//   ws[((mat*32 + kc*8 + nt) * 32 + lane) * 16 + j]
// so the main kernel's per-lane B operand is one contiguous 32-byte load.
// ---------------------------------------------------------------------------
__global__ void prep_w_kernel(const float* __restrict__ wr_u,
                              const float* __restrict__ wr_l,
                              const float* __restrict__ wi_u,
                              const float* __restrict__ wi_l,
                              unsigned short* __restrict__ ws_raw)
{
    __bf16* ws = (__bf16*)ws_raw;
    int tid  = blockIdx.x * blockDim.x + threadIdx.x;   // 0 .. 65535
    int j    = tid & 15;
    int lane = (tid >> 4) & 31;
    int f    = (tid >> 9) & 31;
    int mat  = tid >> 14;
    int kc = f >> 3, nt = f & 7;
    int k = kc * 32 + (lane >> 4) * 8 + (j & 7) + (j >> 3) * 16;
    int n = nt * 16 + (lane & 15);
    const float* W = (mat == 0) ? wr_u : (mat == 1) ? wr_l : (mat == 2) ? wi_u : wi_l;
    ws[tid] = (__bf16)W[n * N_SUB + k];
}

// ---------------------------------------------------------------------------
// Main kernel: one workgroup = 16 batch rows, full 1024-point IFFT.
// ---------------------------------------------------------------------------
union SmemU {
    __bf16 inBf[2 * N_SLOT * TM * N_SUB];   // 64 KB : [comp][slot][row][k] (phase 1/2)
    float  X[2 * N_SLOT * TM * N_SUB];      // 128 KB: Xr then Xi, [slot][row][r] (phase 2/3)
};

__global__ __launch_bounds__(THREADS)
void ifft1024_wmma_kernel(const float* __restrict__ re,
                          const float* __restrict__ im,
                          const unsigned short* __restrict__ ws_raw,
                          float* __restrict__ outR,
                          float* __restrict__ outI)
{
    __shared__ SmemU sm;

    const int  tid     = threadIdx.x;
    const long rowBase = (long)blockIdx.x * TM;

    // ---------------- Phase 1: load + bit-reversal permute + fp32->bf16 ----
    // One "group" = 8 consecutive floats of one row = all 8 slot-phases of one k.
    // 2 comps * 16 rows * 128 k = 4096 groups; 16 per thread.
    for (int i = 0; i < 16; ++i) {
        int g    = i * THREADS + tid;
        int comp = g >> 11;
        int rem  = g & 2047;
        int row  = rem >> 7;
        int k    = rem & 127;
        const float* src = (comp ? im : re) + (rowBase + row) * N_FFT + k * 8;
        float4 x0 = *(const float4*)(src);
        float4 x1 = *(const float4*)(src + 4);
        float v[8] = {x0.x, x0.y, x0.z, x0.w, x1.x, x1.y, x1.z, x1.w};
#pragma unroll
        for (int p = 0; p < 8; ++p) {
            int s = bitrev3(p);   // bitrev is its own inverse
            sm.inBf[(((comp << 3) + s) * TM + row) * N_SUB + k] = (__bf16)v[p];
        }
    }
    __syncthreads();

    // ---------------- Phase 2: complex split-bf16 GEMM via WMMA ------------
    const int wave = tid >> 5;      // slot this wave owns (M-tile)
    const int lane = tid & 31;

    v8f accR[8], accI[8];
#pragma unroll
    for (int nt = 0; nt < 8; ++nt) {
#pragma unroll
        for (int e = 0; e < 8; ++e) { accR[nt][e] = 0.f; accI[nt][e] = 0.f; }
    }

    // A-fragment addressing (16-bit A 16x32 layout): lane<16 -> K 0..7 / 16..23,
    // lane>=16 -> K 8..15 / 24..31, row m = lane&15.
    const int m    = lane & 15;
    const int koff = (lane >> 4) * 8;
    const __bf16* baseR = &sm.inBf[((0 * N_SLOT + wave) * TM + m) * N_SUB];
    const __bf16* baseI = &sm.inBf[((1 * N_SLOT + wave) * TM + m) * N_SUB];
    const v16bf*  wfrag = (const v16bf*)ws_raw;

    union AF { v16bf v; v8bf h[2]; };

    for (int kc = 0; kc < 4; ++kc) {
        int k0 = kc * 32 + koff;
        AF ar, ai, ain;
        ar.h[0] = *(const v8bf*)(baseR + k0);
        ar.h[1] = *(const v8bf*)(baseR + k0 + 16);
        ai.h[0] = *(const v8bf*)(baseI + k0);
        ai.h[1] = *(const v8bf*)(baseI + k0 + 16);
        // -ip fragment (bf16 WMMA has no A-neg modifier; flip sign bits)
        v8u t = *(v8u*)&ai.v;
#pragma unroll
        for (int e = 0; e < 8; ++e) t[e] ^= 0x80008000u;
        ain.v = *(v16bf*)&t;

#pragma unroll
        for (int nt = 0; nt < 8; ++nt) {
            int f = kc * 8 + nt;
            v16bf bru = wfrag[(0 * 32 + f) * 32 + lane];
            v16bf brl = wfrag[(1 * 32 + f) * 32 + lane];
            v16bf biu = wfrag[(2 * 32 + f) * 32 + lane];
            v16bf bil = wfrag[(3 * 32 + f) * 32 + lane];
            // yr += rp*(wr_u + wr_l) - ip*(wi_u + wi_l)
            accR[nt] = __builtin_amdgcn_wmma_f32_16x16x32_bf16(false, ar.v,  false, bru, (short)0, accR[nt], false, false);
            accR[nt] = __builtin_amdgcn_wmma_f32_16x16x32_bf16(false, ar.v,  false, brl, (short)0, accR[nt], false, false);
            accR[nt] = __builtin_amdgcn_wmma_f32_16x16x32_bf16(false, ain.v, false, biu, (short)0, accR[nt], false, false);
            accR[nt] = __builtin_amdgcn_wmma_f32_16x16x32_bf16(false, ain.v, false, bil, (short)0, accR[nt], false, false);
            // yi += rp*(wi_u + wi_l) + ip*(wr_u + wr_l)
            accI[nt] = __builtin_amdgcn_wmma_f32_16x16x32_bf16(false, ar.v,  false, biu, (short)0, accI[nt], false, false);
            accI[nt] = __builtin_amdgcn_wmma_f32_16x16x32_bf16(false, ar.v,  false, bil, (short)0, accI[nt], false, false);
            accI[nt] = __builtin_amdgcn_wmma_f32_16x16x32_bf16(false, ai.v,  false, bru, (short)0, accI[nt], false, false);
            accI[nt] = __builtin_amdgcn_wmma_f32_16x16x32_bf16(false, ai.v,  false, brl, (short)0, accI[nt], false, false);
        }
    }
    __syncthreads();   // all WMMA input reads done; safe to overwrite LDS

    // Spill C fragments to LDS (C layout: VGPR j -> M=j (lanes 0-15) / M=j+8).
    float* Xr = sm.X;
    float* Xi = sm.X + N_SLOT * TM * N_SUB;
    {
        int mrow = (lane >> 4) * 8;
        int colb = lane & 15;
#pragma unroll
        for (int nt = 0; nt < 8; ++nt) {
#pragma unroll
            for (int j = 0; j < 8; ++j) {
                int mm = mrow + j;
                Xr[(wave * TM + mm) * N_SUB + nt * 16 + colb] = accR[nt][j];
                Xi[(wave * TM + mm) * N_SUB + nt * 16 + colb] = accI[nt][j];
            }
        }
    }
    __syncthreads();

    // ---------------- Phase 3: closed-form radix-2 combine -----------------
    // Y[m][c] = (1/8) * sum_s exp(+2*pi*i*c*s/1024) * X_s[m][c & 127]
    const float theta_unit = 6.28318530717958647692f / (float)N_FFT;
    for (int q = 0; q < N_FFT / THREADS; ++q) {
        int c = q * THREADS + tid;
        int r = c & (N_SUB - 1);
        float th = theta_unit * (float)c;
        float sn, cs;
        __sincosf(th, &sn, &cs);
        float cr[8], ci[8];
        cr[0] = 1.f; ci[0] = 0.f;
#pragma unroll
        for (int s = 1; s < 8; ++s) {   // incremental rotation by e^{i*th}
            cr[s] = cr[s - 1] * cs - ci[s - 1] * sn;
            ci[s] = cr[s - 1] * sn + ci[s - 1] * cs;
        }
        for (int mm = 0; mm < TM; ++mm) {
            float yr = 0.f, yi = 0.f;
#pragma unroll
            for (int s = 0; s < 8; ++s) {
                float xr = Xr[(s * TM + mm) * N_SUB + r];
                float xi = Xi[(s * TM + mm) * N_SUB + r];
                yr += cr[s] * xr - ci[s] * xi;
                yi += cr[s] * xi + ci[s] * xr;
            }
            long o = (rowBase + mm) * N_FFT + c;
            outR[o] = yr * 0.125f;
            outI[o] = yi * 0.125f;
        }
    }
}

// ---------------------------------------------------------------------------
extern "C" void kernel_launch(void* const* d_in, const int* in_sizes, int n_in,
                              void* d_out, int out_size, void* d_ws, size_t ws_size,
                              hipStream_t stream)
{
    const float* re   = (const float*)d_in[0];
    const float* im   = (const float*)d_in[1];
    const float* wr_u = (const float*)d_in[2];
    const float* wr_l = (const float*)d_in[3];
    const float* wi_u = (const float*)d_in[4];
    const float* wi_l = (const float*)d_in[5];

    unsigned short* ws = (unsigned short*)d_ws;   // 128 KB of bf16 fragments
    const int batch = in_sizes[0] / N_FFT;

    // Pack W matrices into WMMA-ready bf16 fragments (tiny: 65536 elements).
    prep_w_kernel<<<(4 * N_SUB * N_SUB) / 256, 256, 0, stream>>>(wr_u, wr_l, wi_u, wi_l, ws);

    float* outR = (float*)d_out;
    float* outI = outR + (long)batch * N_FFT;
    ifft1024_wmma_kernel<<<batch / TM, THREADS, 0, stream>>>(re, im, ws, outR, outI);
}
